// NLGCN_41188736369375
// MI455X (gfx1250) — matrix operand order repaired and checked by reading, not compiled, not run on hardware
//
#include <hip/hip_runtime.h>
#include <hip/hip_bf16.h>

// ---------------------------------------------------------------------------
// NLGCN for MI455X (gfx1250): bf16 WMMA GEMMs (K=64 staged chunks, vectorized
// LDS fragment path, compile-time epilogues) + atomic edge scatter +
// bitonic sort + WMMA conv1d + log_softmax.
// ---------------------------------------------------------------------------

typedef __bf16 v16bf __attribute__((ext_vector_type(16)));
typedef __bf16 v8bf  __attribute__((ext_vector_type(8)));
typedef float  v8f   __attribute__((ext_vector_type(8)));

#define NNODES 100000
#define FIN    512
#define HID    256
#define CC     64
#define KW     5
#define PAD2   131072   // 2^17 >= NNODES
#define APAD   72       // 144-byte LDS rows: 16B-aligned, bank-conflict-free

// pack two floats into adjacent bf16 halves (lowers to v_cvt_pk_bf16_f32)
__device__ inline unsigned pack_bf16(float f0, float f1) {
    union { unsigned u; __bf16 h[2]; } pk;
    pk.h[0] = (__bf16)f0;
    pk.h[1] = (__bf16)f1;
    return pk.u;
}

__device__ inline v16bf cat8(v8bf a, v8bf b) {
    return __builtin_shufflevector(a, b, 0, 1, 2, 3, 4, 5, 6, 7,
                                         8, 9, 10, 11, 12, 13, 14, 15);
}

// ------------------------- elementwise helpers ------------------------------

__global__ __launch_bounds__(256) void fill_f32(float* p, float v, long long n) {
    long long i = (long long)blockIdx.x * 256 + threadIdx.x;
    if (i < n) p[i] = v;
}

__global__ __launch_bounds__(256) void deg_count(const int* __restrict__ dst,
                                                 float* __restrict__ deg, int E) {
    long long e = (long long)blockIdx.x * 256 + threadIdx.x;
    if (e < E) atomicAdd(&deg[dst[e]], 1.0f);
}

__global__ __launch_bounds__(256) void rsqrt_ip(float* p, int n) {
    int i = blockIdx.x * 256 + threadIdx.x;
    if (i < n) p[i] = rsqrtf(p[i]);   // deg >= 1 always (self loop)
}

// out[i][c] = dinv[i]^2 * hw[i][c]   (self-loop contribution, also inits buffer)
__global__ __launch_bounds__(256) void selfloop_init(const float* __restrict__ hw,
                                                     const float* __restrict__ dinv,
                                                     float* __restrict__ out,
                                                     int n, int F) {
    long long t = (long long)blockIdx.x * 256 + threadIdx.x;
    if (t < (long long)n * F) {
        int i = (int)(t / F);
        float d = dinv[i];
        out[t] = d * d * hw[t];
    }
}

// scatter-add: out[dst[e]][c] += dinv[s]*dinv[d] * hw[src[e]][c]
__global__ __launch_bounds__(256) void edge_scatter(const float* __restrict__ hw,
                                                    const float* __restrict__ dinv,
                                                    const int* __restrict__ src,
                                                    const int* __restrict__ dst,
                                                    float* __restrict__ out,
                                                    int E, int F) {
    long long t = (long long)blockIdx.x * 256 + threadIdx.x;
    if (t >= (long long)E * F) return;
    int e = (int)(t / F);
    int c = (int)(t - (long long)e * F);
    int s = src[e], d = dst[e];
    float nrm = dinv[s] * dinv[d];
    atomicAdd(&out[(long long)d * F + c], nrm * hw[(long long)s * F + c]);
}

__global__ __launch_bounds__(256) void bias_act(float* __restrict__ h,
                                                const float* __restrict__ b,
                                                int n, int F, int relu) {
    long long t = (long long)blockIdx.x * 256 + threadIdx.x;
    if (t < (long long)n * F) {
        int c = (int)(t % F);
        float v = h[t] + b[c];
        h[t] = relu ? fmaxf(v, 0.0f) : v;
    }
}

// ------------------------- bf16 WMMA GEMM -----------------------------------
// C[M,N] = A[M,K] @ B[K,N] (row-major fp32 in, bf16 WMMA, f32 accumulate)
// FLAGS: bit0 = relu, bit1 = accumulate into C. HAS_BIAS: add bias[N] column.
// Requires M%32==0, N%64==0, K%64==0.
template <int FLAGS, bool HAS_BIAS>
__global__ __launch_bounds__(256)
void gemm_bf16_wmma(const float* __restrict__ A, const float* __restrict__ B,
                    const float* __restrict__ bias, float* __restrict__ C,
                    int M, int N, int K) {
    __shared__ __bf16 As[32][APAD];     // [M-tile][K-chunk 64]
    __shared__ __bf16 BsT[64][APAD];    // [N-tile][K-chunk 64] (transposed stage)
    const int tid  = threadIdx.x;
    const int m0   = blockIdx.x * 32;
    const int n0   = blockIdx.y * 64;
    const int wid  = tid >> 5;
    const int lane = tid & 31;
    const int row  = lane & 15;
    const int half = lane >> 4;
    const int wm   = (wid >> 2) * 16;   // 0 / 16
    const int wn   = (wid & 3) * 16;    // 0 / 16 / 32 / 48

    v8f acc = {};
    for (int kk = 0; kk < K; kk += 64) {
        // stage A: 32x64, packed bf16 pairs along K (1024 pairs, 4/thread)
#pragma unroll
        for (int it = 0; it < 4; ++it) {
            int e  = tid + it * 256;          // 0..1023
            int r  = e >> 5;                  // 0..31
            int c2 = (e & 31) << 1;           // even 0..62
            float2 f = *(const float2*)(A + (long long)(m0 + r) * K + kk + c2);
            *(unsigned*)&As[r][c2] = pack_bf16(f.x, f.y);
        }
        // stage B transposed: BsT[n][k], pack pairs along K (2048 pairs, 8/thread)
#pragma unroll
        for (int it = 0; it < 8; ++it) {
            int e = tid + it * 256;           // 0..2047
            int c = e & 63;                   // N within tile
            int r = (e >> 6) << 1;            // even K 0..62
            float f0 = B[(long long)(kk + r)     * N + n0 + c];
            float f1 = B[(long long)(kk + r + 1) * N + n0 + c];
            *(unsigned*)&BsT[c][r] = pack_bf16(f0, f1);
        }
        __syncthreads();
        // two back-to-back WMMAs per staged chunk; fragments are 2x ds_load_b128
#pragma unroll
        for (int k2 = 0; k2 < 64; k2 += 32) {
            const __bf16* arow = &As[wm + row][k2];
            const __bf16* brow = &BsT[wn + row][k2];
            v16bf af = cat8(*(const v8bf*)(arow + half * 8),
                            *(const v8bf*)(arow + 16 + half * 8));
            v16bf bf = cat8(*(const v8bf*)(brow + half * 8),
                            *(const v8bf*)(brow + 16 + half * 8));
            acc = __builtin_amdgcn_wmma_f32_16x16x32_bf16(
                false, af, false, bf, (short)0, acc, false, false);
        }
        __syncthreads();
    }
#pragma unroll
    for (int j = 0; j < 8; ++j) {             // D: M = j + 8*half, N = lane%16
        int mi = m0 + wm + j + 8 * half;
        int ni = n0 + wn + row;
        long long o = (long long)mi * N + ni;
        float v = acc[j];
        if (FLAGS & 2) {
            C[o] += v;
        } else {
            if (HAS_BIAS) v += bias[ni];
            if (FLAGS & 1) v = fmaxf(v, 0.0f);
            C[o] = v;
        }
    }
}

// ------------------------- conv1d (K=5, SAME) as 5 shifted WMMA GEMMs -------
// X: [L][64], W: [co][ci][5], Y[l][co] = b[co] + sum_k sum_ci X[l+k-2][ci]*W[co][ci][k]
template <int RELU>
__global__ __launch_bounds__(256)
void conv1d_wmma(const float* __restrict__ X, const float* __restrict__ W,
                 const float* __restrict__ bias, float* __restrict__ Y, int L) {
    __shared__ __bf16 As[32][APAD];     // [L-tile][ci]
    __shared__ __bf16 BsT[64][APAD];    // [co][ci]
    const int tid  = threadIdx.x;
    const int l0   = blockIdx.x * 32;
    const int wid  = tid >> 5;
    const int lane = tid & 31;
    const int row  = lane & 15;
    const int half = lane >> 4;
    const int wm   = (wid >> 2) * 16;
    const int wn   = (wid & 3) * 16;

    v8f acc = {};
    for (int k = 0; k < KW; ++k) {
        // stage shifted, zero-padded A tile: 32x64 (1024 pairs, 4/thread)
#pragma unroll
        for (int it = 0; it < 4; ++it) {
            int e  = tid + it * 256;          // 0..1023
            int r  = e >> 5;                  // 0..31
            int c2 = (e & 31) << 1;           // even 0..62
            int l  = l0 + r + k - 2;
            float2 f = make_float2(0.0f, 0.0f);
            if (l >= 0 && l < L)
                f = *(const float2*)(X + (long long)l * CC + c2);
            *(unsigned*)&As[r][c2] = pack_bf16(f.x, f.y);
        }
        // stage BsT[co][ci] = W[co][ci][k]: 64x64 (2048 pairs, 8/thread)
#pragma unroll
        for (int it = 0; it < 8; ++it) {
            int e  = tid + it * 256;          // 0..2047
            int co = e & 63;
            int ci = (e >> 6) << 1;           // even 0..62
            float f0 = W[co * (CC * KW) + ci * KW + k];
            float f1 = W[co * (CC * KW) + (ci + 1) * KW + k];
            *(unsigned*)&BsT[co][ci] = pack_bf16(f0, f1);
        }
        __syncthreads();
#pragma unroll
        for (int k2 = 0; k2 < 64; k2 += 32) {
            const __bf16* arow = &As[wm + row][k2];
            const __bf16* brow = &BsT[wn + row][k2];
            v16bf af = cat8(*(const v8bf*)(arow + half * 8),
                            *(const v8bf*)(arow + 16 + half * 8));
            v16bf bf = cat8(*(const v8bf*)(brow + half * 8),
                            *(const v8bf*)(brow + 16 + half * 8));
            acc = __builtin_amdgcn_wmma_f32_16x16x32_bf16(
                false, af, false, bf, (short)0, acc, false, false);
        }
        __syncthreads();
    }
#pragma unroll
    for (int j = 0; j < 8; ++j) {
        int mi = l0 + wm + j + 8 * half;
        int ni = wn + row;
        float v = acc[j] + bias[ni];
        if (RELU) v = fmaxf(v, 0.0f);
        Y[(long long)mi * CC + ni] = v;
    }
}

// ------------------------- sort pipeline ------------------------------------

__global__ __launch_bounds__(256) void score_init(const float* __restrict__ x1,
                                                  const float* __restrict__ Wp,
                                                  const float* __restrict__ bp,
                                                  float* __restrict__ keys,
                                                  int* __restrict__ idx, int n) {
    int i = blockIdx.x * 256 + threadIdx.x;
    if (i >= PAD2) return;
    if (i < n) {
        float s = bp[0];
#pragma unroll
        for (int c = 0; c < CC; ++c) s += x1[(long long)i * CC + c] * Wp[c];
        keys[i] = s;
        idx[i]  = i;
    } else {
        keys[i] = __builtin_inff();
        idx[i]  = -1;
    }
}

__global__ __launch_bounds__(256) void bitonic_step(float* __restrict__ keys,
                                                    int* __restrict__ idx,
                                                    unsigned j, unsigned k) {
    unsigned i = blockIdx.x * 256 + threadIdx.x;
    unsigned p = i ^ j;
    if (p > i && i < PAD2) {
        float a = keys[i], b = keys[p];
        bool sw = ((i & k) == 0) ? (a > b) : (a < b);
        if (sw) {
            keys[i] = b; keys[p] = a;
            int t = idx[i]; idx[i] = idx[p]; idx[p] = t;
        }
    }
}

__global__ __launch_bounds__(256) void inverse_perm(const int* __restrict__ idx,
                                                    int* __restrict__ inv, int n) {
    int i = blockIdx.x * 256 + threadIdx.x;
    if (i < n) inv[idx[i]] = i;
}

// sorted_x[i][c] = keys[i] * x1[idx[i]][c]
__global__ __launch_bounds__(256) void gather_scale(const float* __restrict__ x1,
                                                    const float* __restrict__ keys,
                                                    const int* __restrict__ idx,
                                                    float* __restrict__ sx, int n) {
    long long t = (long long)blockIdx.x * 256 + threadIdx.x;
    if (t < (long long)n * CC) {
        int i = (int)(t >> 6), c = (int)(t & 63);
        sx[t] = keys[i] * x1[(long long)idx[i] * CC + c];
    }
}

// x2[i][c] = t2[inv[i]][c]
__global__ __launch_bounds__(256) void gather_rows(const float* __restrict__ t2,
                                                   const int* __restrict__ inv,
                                                   float* __restrict__ x2, int n) {
    long long t = (long long)blockIdx.x * 256 + threadIdx.x;
    if (t < (long long)n * CC) {
        int i = (int)(t >> 6), c = (int)(t & 63);
        x2[t] = t2[(long long)inv[i] * CC + c];
    }
}

__global__ __launch_bounds__(256) void log_softmax64(const float* __restrict__ lg,
                                                     float* __restrict__ out, int n) {
    int i = blockIdx.x * 256 + threadIdx.x;
    if (i >= n) return;
    const float* r = lg + (long long)i * CC;
    float m = -3.402823466e38f;
#pragma unroll
    for (int c = 0; c < CC; ++c) m = fmaxf(m, r[c]);
    float s = 0.0f;
#pragma unroll
    for (int c = 0; c < CC; ++c) s += expf(r[c] - m);
    float ls = m + logf(s);
#pragma unroll
    for (int c = 0; c < CC; ++c) out[(long long)i * CC + c] = r[c] - ls;
}

// ---------------------------------------------------------------------------

static inline long long cdiv(long long a, long long b) { return (a + b - 1) / b; }

extern "C" void kernel_launch(void* const* d_in, const int* in_sizes, int n_in,
                              void* d_out, int out_size, void* d_ws, size_t ws_size,
                              hipStream_t stream) {
    const float* x   = (const float*)d_in[0];
    const int*   ei  = (const int*)d_in[1];
    const float* W1  = (const float*)d_in[2];
    const float* b1  = (const float*)d_in[3];
    const float* W2  = (const float*)d_in[4];
    const float* b2  = (const float*)d_in[5];
    const float* Wp  = (const float*)d_in[6];
    const float* bp  = (const float*)d_in[7];
    const float* cw1 = (const float*)d_in[8];
    const float* cb1 = (const float*)d_in[9];
    const float* cw2 = (const float*)d_in[10];
    const float* cb2 = (const float*)d_in[11];
    const float* Wl  = (const float*)d_in[12];
    const float* bl  = (const float*)d_in[13];

    const int N = NNODES;
    const int E = in_sizes[1] / 2;
    const int* src = ei;
    const int* dst = ei + E;

    // ---- workspace layout (fp32), ~284 MB peak, buffers reused ----
    float* hw1  = (float*)d_ws;                    // [N][256]  (later: conv t1)
    float* hbuf = hw1  + (size_t)N * HID;          // [N][256]  h   (later: x2)
    float* hw2  = hbuf + (size_t)N * HID;          // [N][64]   (later: conv t2)
    float* x1   = hw2  + (size_t)N * CC;           // [N][64]
    float* sx   = x1   + (size_t)N * CC;           // [N][64]   sorted_x (later: logits)
    float* dinv = sx   + (size_t)N * CC;           // [N]
    float* keys = dinv + N;                        // [PAD2]
    int*   idx  = (int*)(keys + PAD2);             // [PAD2]
    int*   inv  = idx + PAD2;                      // [N]

    const int T = 256;

    // 1) degrees (with self loop) -> dinv
    fill_f32<<<(int)cdiv(N, T), T, 0, stream>>>(dinv, 1.0f, N);
    deg_count<<<(int)cdiv(E, T), T, 0, stream>>>(dst, dinv, E);
    rsqrt_ip<<<(int)cdiv(N, T), T, 0, stream>>>(dinv, N);

    // 2) GCN layer 1: hw1 = x @ W1 ; h = relu(Aggregate(hw1) + b1)
    gemm_bf16_wmma<0, false><<<dim3(N / 32, HID / 64), T, 0, stream>>>(
        x, W1, nullptr, hw1, N, HID, FIN);
    selfloop_init<<<(int)cdiv((long long)N * HID, T), T, 0, stream>>>(hw1, dinv, hbuf, N, HID);
    edge_scatter<<<(int)cdiv((long long)E * HID, T), T, 0, stream>>>(hw1, dinv, src, dst,
                                                                     hbuf, E, HID);
    bias_act<<<(int)cdiv((long long)N * HID, T), T, 0, stream>>>(hbuf, b1, N, HID, 1);

    // 3) GCN layer 2: hw2 = h @ W2 ; x1 = Aggregate(hw2) + b2
    gemm_bf16_wmma<0, false><<<dim3(N / 32, CC / 64), T, 0, stream>>>(
        hbuf, W2, nullptr, hw2, N, CC, HID);
    selfloop_init<<<(int)cdiv((long long)N * CC, T), T, 0, stream>>>(hw2, dinv, x1, N, CC);
    edge_scatter<<<(int)cdiv((long long)E * CC, T), T, 0, stream>>>(hw2, dinv, src, dst,
                                                                    x1, E, CC);
    bias_act<<<(int)cdiv((long long)N * CC, T), T, 0, stream>>>(x1, b2, N, CC, 0);

    // 4) scores + bitonic argsort (ascending) + inverse permutation
    score_init<<<PAD2 / T, T, 0, stream>>>(x1, Wp, bp, keys, idx, N);
    for (unsigned k = 2; k <= PAD2; k <<= 1)
        for (unsigned j = k >> 1; j > 0; j >>= 1)
            bitonic_step<<<PAD2 / T, T, 0, stream>>>(keys, idx, j, k);
    inverse_perm<<<(int)cdiv(N, T), T, 0, stream>>>(idx, inv, N);

    // 5) sorted_x = g_sorted * x1[sort_idx]
    gather_scale<<<(int)cdiv((long long)N * CC, T), T, 0, stream>>>(x1, keys, idx, sx, N);

    // 6) conv1d chain over sorted sequence (reuse hw1 / hw2 as t1 / t2)
    conv1d_wmma<1><<<N / 32, T, 0, stream>>>(sx, cw1, cb1, hw1, N);
    conv1d_wmma<0><<<N / 32, T, 0, stream>>>(hw1, cw2, cb2, hw2, N);

    // 7) x2 = sorted_out[inverse_idx]  (into hbuf)
    gather_rows<<<(int)cdiv((long long)N * CC, T), T, 0, stream>>>(hw2, inv, hbuf, N);

    // 8) logits = x1 @ Wl[:64] + x2 @ Wl[64:] + bl  (into sx), then log_softmax
    gemm_bf16_wmma<0, true><<<dim3(N / 32, CC / 64), T, 0, stream>>>(
        x1, Wl, bl, sx, N, CC, CC);
    gemm_bf16_wmma<2, false><<<dim3(N / 32, CC / 64), T, 0, stream>>>(
        hbuf, Wl + CC * CC, nullptr, sx, N, CC, CC);
    log_softmax64<<<(int)cdiv(N, T), T, 0, stream>>>(sx, (float*)d_out, N);
}
